// CTCGuidedRefiner_63462436765736
// MI455X (gfx1250) — matrix operand order repaired and checked by reading, not compiled, not run on hardware
//
#include <hip/hip_runtime.h>
#include <hip/hip_bf16.h>
#include <math.h>

// ---------------------------------------------------------------------------
// CTCGuidedRefiner on MI455X (gfx1250): bf16 WMMA GEMMs + VALU band-attn.
// ---------------------------------------------------------------------------

typedef __attribute__((ext_vector_type(16))) __bf16 v16bf;
typedef __attribute__((ext_vector_type(8)))  __bf16 bf16x8;
typedef __attribute__((ext_vector_type(8)))  float  v8f;

constexpr int Bc  = 2;
constexpr int Tc  = 2048;
constexpr int Dc  = 512;
constexpr int Fc  = 2048;   // 4*D FFN hidden
constexpr int Vc  = 1024;
constexpr int Hc  = 8;
constexpr int HDc = 64;
constexpr int Mc  = Bc * Tc;  // 4096 rows

#define DEVFN __device__ __forceinline__

// ---------------------------------------------------------------------------
// CTC stats: per (b,t): H_k, A_k for k=0,1; A_g = clip(max), H_g = mean.
// One 256-thread block per row, streaming V=1024 logits per speaker.
// ---------------------------------------------------------------------------
__global__ __launch_bounds__(256)
void k_ctc_stats(const float* __restrict__ logits, const int* __restrict__ blankPtr,
                 float* __restrict__ Ak, float* __restrict__ Ag, float* __restrict__ Hg)
{
    __shared__ float red[256];
    const int row = blockIdx.x;          // b*T + t
    const int tid = threadIdx.x;
    const int blank = blankPtr[0];
    float Av[2], Hv[2];
    for (int k = 0; k < 2; ++k) {
        const float* lp = logits + ((size_t)k * Mc + row) * Vc;
        float m = -1e30f;
        for (int v = tid; v < Vc; v += 256) m = fmaxf(m, lp[v]);
        red[tid] = m; __syncthreads();
        for (int s = 128; s > 0; s >>= 1) { if (tid < s) red[tid] = fmaxf(red[tid], red[tid + s]); __syncthreads(); }
        m = red[0]; __syncthreads();
        float se = 0.f, sl = 0.f;
        for (int v = tid; v < Vc; v += 256) {
            float l = lp[v];
            float e = __expf(l - m);
            se += e; sl += e * l;
        }
        red[tid] = se; __syncthreads();
        for (int s = 128; s > 0; s >>= 1) { if (tid < s) red[tid] += red[tid + s]; __syncthreads(); }
        se = red[0]; __syncthreads();
        red[tid] = sl; __syncthreads();
        for (int s = 128; s > 0; s >>= 1) { if (tid < s) red[tid] += red[tid + s]; __syncthreads(); }
        sl = red[0]; __syncthreads();
        float logZ = m + __logf(se);
        Hv[k] = logZ - sl / se;                      // -(sum p*logp)
        Av[k] = 1.f - __expf(lp[blank] - logZ);      // 1 - P(blank)
    }
    if (tid == 0) {
        Ak[row]      = Av[0];
        Ak[Mc + row] = Av[1];
        float ag = fmaxf(Av[0], Av[1]);
        Ag[row] = fminf(fmaxf(ag, 0.f), 1.f);
        Hg[row] = 0.5f * (Hv[0] + Hv[1]);
    }
}

// ---------------------------------------------------------------------------
// Router gate: gate = sigmoid(softmax_k(A_k) @ mix_w + mix_b), per (b,t)
// ---------------------------------------------------------------------------
__global__ __launch_bounds__(256)
void k_gate(const float* __restrict__ Ak, const float* __restrict__ mixw,
            const float* __restrict__ mixb, float* __restrict__ gate, int n)
{
    int i = blockIdx.x * 256 + threadIdx.x;
    if (i >= n) return;
    float a0 = Ak[i], a1 = Ak[n + i];
    float m  = fmaxf(a0, a1);
    float e0 = __expf(a0 - m), e1 = __expf(a1 - m);
    float inv = 1.f / (e0 + e1);
    float z = (e0 * inv) * mixw[0] + (e1 * inv) * mixw[1] + mixb[0];
    gate[i] = 1.f / (1.f + __expf(-z));
}

// ---------------------------------------------------------------------------
// LPF center coefficient: c = softmax(silu(g@w1+b1)@w2+b2)[4], per (b,t).
// Block 256 per row; 9 block reductions.
// ---------------------------------------------------------------------------
__global__ __launch_bounds__(256)
void k_lpf_coeff(const float* __restrict__ Ag, const float* __restrict__ Hg,
                 const float* __restrict__ w1, const float* __restrict__ b1,
                 const float* __restrict__ w2, const float* __restrict__ b2,
                 float* __restrict__ cOut)
{
    __shared__ float red[256];
    __shared__ float logit[9];
    const int row = blockIdx.x, tid = threadIdx.x;
    float a = Ag[row], hg = Hg[row];
    float part[9];
    #pragma unroll
    for (int j = 0; j < 9; ++j) part[j] = 0.f;
    for (int ch = tid; ch < Dc; ch += 256) {
        float z = a * w1[ch] + hg * w1[Dc + ch] + b1[ch];
        float hs = z / (1.f + __expf(-z));           // silu
        #pragma unroll
        for (int j = 0; j < 9; ++j) part[j] += hs * w2[ch * 9 + j];
    }
    for (int j = 0; j < 9; ++j) {
        red[tid] = part[j]; __syncthreads();
        for (int s = 128; s > 0; s >>= 1) { if (tid < s) red[tid] += red[tid + s]; __syncthreads(); }
        if (tid == 0) logit[j] = red[0] + b2[j];
        __syncthreads();
    }
    if (tid == 0) {
        float m = -1e30f;
        for (int j = 0; j < 9; ++j) m = fmaxf(m, logit[j]);
        float s = 0.f, p4 = 0.f;
        for (int j = 0; j < 9; ++j) { float e = __expf(logit[j] - m); s += e; if (j == 4) p4 = e; }
        cOut[row] = p4 / s;
    }
}

// ---------------------------------------------------------------------------
// LPF apply: yo = y + c*y + (1-c)*depthwise_conv9(y)
// ---------------------------------------------------------------------------
__global__ __launch_bounds__(256)
void k_lpf_apply(const float* __restrict__ y, const float* __restrict__ dw,
                 const float* __restrict__ c, float* __restrict__ yo, int n)
{
    int i = blockIdx.x * 256 + threadIdx.x;
    if (i >= n) return;
    int d = i & (Dc - 1);
    int row = i >> 9;
    int t = row & (Tc - 1);
    int b = row >> 11;
    float yc = 0.f;
    #pragma unroll
    for (int u = 0; u < 9; ++u) {
        int tt = t + u - 4;
        if (tt >= 0 && tt < Tc)
            yc += dw[d * 9 + u] * y[(((size_t)(b * Tc + tt)) << 9) + d];
    }
    float cc = c[row], yv = y[i];
    yo[i] = yv + cc * yv + (1.f - cc) * yc;
}

// ---------------------------------------------------------------------------
// LayerNorm over D=512, one wave32 per row (16 elems/lane, shfl reductions).
// Writes fp32 and/or bf16 (feeds WMMA GEMMs directly).
// ---------------------------------------------------------------------------
__global__ __launch_bounds__(256)
void k_layernorm(const float* __restrict__ x, const float* __restrict__ g,
                 const float* __restrict__ bb, float* __restrict__ outF,
                 __bf16* __restrict__ outB)
{
    const int wave = threadIdx.x >> 5, lane = threadIdx.x & 31;
    const int row = blockIdx.x * 8 + wave;
    const float* xp = x + (size_t)row * Dc;
    float v[16];
    float s = 0.f;
    #pragma unroll
    for (int j = 0; j < 16; ++j) { v[j] = xp[lane + 32 * j]; s += v[j]; }
    for (int off = 16; off > 0; off >>= 1) s += __shfl_xor(s, off, 32);
    float mean = s * (1.f / 512.f);
    float q = 0.f;
    #pragma unroll
    for (int j = 0; j < 16; ++j) { float d = v[j] - mean; q += d * d; }
    for (int off = 16; off > 0; off >>= 1) q += __shfl_xor(q, off, 32);
    float inv = rsqrtf(q * (1.f / 512.f) + 1e-5f);
    #pragma unroll
    for (int j = 0; j < 16; ++j) {
        int col = lane + 32 * j;
        float val = (v[j] - mean) * inv * g[col] + bb[col];
        size_t idx = (size_t)row * Dc + col;
        if (outF) outF[idx] = val;
        if (outB) outB[idx] = (__bf16)val;
    }
}

// ---------------------------------------------------------------------------
// Elementwise helpers
// ---------------------------------------------------------------------------
__global__ __launch_bounds__(256)
void k_f32_to_bf16(const float* __restrict__ in, __bf16* __restrict__ out, int n)
{
    int i = blockIdx.x * 256 + threadIdx.x;
    if (i < n) out[i] = (__bf16)in[i];
}

__global__ __launch_bounds__(256)
void k_scale_gate(const float* __restrict__ y, const float* __restrict__ gate,
                  __bf16* __restrict__ out, int n)
{
    int i = blockIdx.x * 256 + threadIdx.x;
    if (i >= n) return;
    int row = i >> 9;
    out[i] = (__bf16)(y[i] * (0.5f + 0.5f * gate[row]));
}

// Weight transpose+convert: w[K,N] f32 -> wt[N,K] bf16 (so B fragments are
// contiguous 16B loads along K). Weights live in L2 (192MB), this is cheap.
__global__ __launch_bounds__(256)
void k_transpose_bf16(const float* __restrict__ w, __bf16* __restrict__ wt, int K, int N)
{
    int i = blockIdx.x * 256 + threadIdx.x;
    if (i >= K * N) return;
    int n = i / K, k = i - n * K;
    wt[i] = (__bf16)w[(size_t)k * N + n];
}

// ---------------------------------------------------------------------------
// bf16 WMMA GEMM: out[M,N] = act(A[M,K] @ Wt[N,K]^T + bias) (+ res)
// Block 256 = 8 waves tiling 128(m) x 64(n); each wave: 32x32 via 2x2
// v_wmma_f32_16x16x32_bf16 fragments, K-step 32.
// Fragment layouts per CDNA5 ISA 7.12.2:
//   A (16x32, 16-bit): lane half h = lane>>4, row m = lane&15;
//       elem e -> k = 16*(e>>3) + 8*h + (e&7)     => two 16B loads
//   B (32x16): col n = lane&15; elem e -> k = 16*(e>>3) + 8*h + (e&7)  => two 16B loads (Wt)
//   C/D (16x16 f32): col = lane&15, vgpr r -> row = 8*h + r
// ---------------------------------------------------------------------------
DEVFN void gemm_emit(float v, float biasv, const float* res,
                     float* outF, __bf16* outB, size_t idx, int act)
{
    v += biasv;
    if (act == 1) v = 0.5f * v * (1.f + erff(v * 0.70710678118654752f));  // exact gelu
    if (res)  v += res[idx];
    if (outF) outF[idx] = v;
    if (outB) outB[idx] = (__bf16)v;
}

DEVFN v16bf cat8(bf16x8 lo, bf16x8 hi)
{
    return __builtin_shufflevector(lo, hi, 0,1,2,3,4,5,6,7,8,9,10,11,12,13,14,15);
}

__global__ __launch_bounds__(256)
void k_gemm_bf16(const __bf16* __restrict__ A, const __bf16* __restrict__ Wt,
                 const float* __restrict__ bias, const float* __restrict__ res,
                 float* __restrict__ outF, __bf16* __restrict__ outB,
                 int M, int N, int K, int act)
{
    const int tid  = threadIdx.x;
    const int wave = tid >> 5, lane = tid & 31;
    const int half = lane >> 4, l16 = lane & 15;
    const int wm = wave >> 1, wn = wave & 1;
    const int m0 = blockIdx.y * 128 + wm * 32;
    const int n0 = blockIdx.x * 64  + wn * 32;

    v8f acc00 = {}, acc01 = {}, acc10 = {}, acc11 = {};

    const __bf16* Ar0 = A  + (size_t)(m0 + l16)      * K;
    const __bf16* Ar1 = A  + (size_t)(m0 + 16 + l16) * K;
    const __bf16* Br0 = Wt + (size_t)(n0 + l16)      * K;
    const __bf16* Br1 = Wt + (size_t)(n0 + 16 + l16) * K;
    const int aoff = 8 * half;       // A: e0..7 at k0+8h, e8..15 at k0+16+8h
    const int boff = 8 * half;       // B: e0..7 at k0+8h, e8..15 at k0+16+8h

    for (int k0 = 0; k0 < K; k0 += 32) {
        v16bf a0 = cat8(*(const bf16x8*)(Ar0 + k0 + aoff),
                        *(const bf16x8*)(Ar0 + k0 + 16 + aoff));
        v16bf a1 = cat8(*(const bf16x8*)(Ar1 + k0 + aoff),
                        *(const bf16x8*)(Ar1 + k0 + 16 + aoff));
        v16bf b0 = cat8(*(const bf16x8*)(Br0 + k0 + boff),
                        *(const bf16x8*)(Br0 + k0 + 16 + boff));
        v16bf b1 = cat8(*(const bf16x8*)(Br1 + k0 + boff),
                        *(const bf16x8*)(Br1 + k0 + 16 + boff));
        acc00 = __builtin_amdgcn_wmma_f32_16x16x32_bf16(false, a0, false, b0, (short)0, acc00, false, false);
        acc01 = __builtin_amdgcn_wmma_f32_16x16x32_bf16(false, a0, false, b1, (short)0, acc01, false, false);
        acc10 = __builtin_amdgcn_wmma_f32_16x16x32_bf16(false, a1, false, b0, (short)0, acc10, false, false);
        acc11 = __builtin_amdgcn_wmma_f32_16x16x32_bf16(false, a1, false, b1, (short)0, acc11, false, false);
    }

    const int gn0 = n0 + l16, gn1 = n0 + 16 + l16;
    const float bias0 = bias ? bias[gn0] : 0.f;
    const float bias1 = bias ? bias[gn1] : 0.f;
    #pragma unroll
    for (int r = 0; r < 8; ++r) {
        int gm0 = m0 + 8 * half + r;
        int gm1 = gm0 + 16;
        gemm_emit(acc00[r], bias0, res, outF, outB, (size_t)gm0 * N + gn0, act);
        gemm_emit(acc01[r], bias1, res, outF, outB, (size_t)gm0 * N + gn1, act);
        gemm_emit(acc10[r], bias0, res, outF, outB, (size_t)gm1 * N + gn0, act);
        gemm_emit(acc11[r], bias1, res, outF, outB, (size_t)gm1 * N + gn1, act);
    }
}

// ---------------------------------------------------------------------------
// Banded MHA core: one wave per (b,h,q). <=49 keys, hd=64.
// Lane j handles keys j, j+32; softmax via shfl; output via LDS-broadcast
// probabilities, each lane produces dims {lane, lane+32}.
// ---------------------------------------------------------------------------
__global__ __launch_bounds__(256)
void k_banded_attn(const float* __restrict__ Q, const float* __restrict__ Kx,
                   const float* __restrict__ Vx, float* __restrict__ O,
                   int band, float scale)
{
    __shared__ float sQ[8][64];
    __shared__ float sP[8][64];
    const int wave = threadIdx.x >> 5, lane = threadIdx.x & 31;
    const int gq = blockIdx.x * 8 + wave;
    const int b = gq / (Hc * Tc);
    const int rem = gq - b * (Hc * Tc);
    const int h = rem / Tc;
    const int t = rem - h * Tc;
    const size_t rowBase = (size_t)(b * Tc + t) * Dc + h * HDc;

    sQ[wave][lane]      = Q[rowBase + lane];
    sQ[wave][lane + 32] = Q[rowBase + lane + 32];
    __syncthreads();

    const int nk = 2 * band + 1;
    float s[2];
    float mx = -1e30f;
    #pragma unroll
    for (int jj = 0; jj < 2; ++jj) {
        int j = lane + jj * 32;
        float val = -1e30f;
        if (j < nk) {
            int kk = t - band + j;
            if (kk >= 0 && kk < Tc) {
                const float* kp = Kx + (size_t)(b * Tc + kk) * Dc + h * HDc;
                float d = 0.f;
                #pragma unroll
                for (int u = 0; u < HDc; ++u) d += sQ[wave][u] * kp[u];
                val = d * scale;
            }
        }
        s[jj] = val;
        mx = fmaxf(mx, val);
    }
    for (int off = 16; off > 0; off >>= 1) mx = fmaxf(mx, __shfl_xor(mx, off, 32));
    float sum = 0.f;
    #pragma unroll
    for (int jj = 0; jj < 2; ++jj) {
        float p = (s[jj] > -1e29f) ? __expf(s[jj] - mx) : 0.f;
        s[jj] = p; sum += p;
    }
    for (int off = 16; off > 0; off >>= 1) sum += __shfl_xor(sum, off, 32);
    float inv = 1.f / sum;
    sP[wave][lane]      = s[0] * inv;
    sP[wave][lane + 32] = s[1] * inv;
    __syncthreads();

    float o0 = 0.f, o1 = 0.f;
    for (int j = 0; j < nk; ++j) {
        int kk = t - band + j;
        if (kk < 0 || kk >= Tc) continue;
        float p = sP[wave][j];
        const float* vp = Vx + (size_t)(b * Tc + kk) * Dc + h * HDc;
        o0 += p * vp[lane];
        o1 += p * vp[lane + 32];
    }
    O[rowBase + lane]      = o0;
    O[rowBase + lane + 32] = o1;
}

// ---------------------------------------------------------------------------
// Host orchestration
// ---------------------------------------------------------------------------
extern "C" void kernel_launch(void* const* d_in, const int* in_sizes, int n_in,
                              void* d_out, int out_size, void* d_ws, size_t ws_size,
                              hipStream_t stream)
{
    (void)in_sizes; (void)n_in; (void)out_size; (void)ws_size;

    const float* x_m    = (const float*)d_in[0];
    const float* logits = (const float*)d_in[1];
    const int*   blank  = (const int*)d_in[2];
    const float* inp_w  = (const float*)d_in[3];
    const float* inp_b  = (const float*)d_in[4];
    const float* oln_g  = (const float*)d_in[5];
    const float* oln_b  = (const float*)d_in[6];
    // Per-layer params, depth-first insertion order; 37 tensors per layer:
    //  0 lpf.w1  1 lpf.b1  2 lpf.w2  3 lpf.b2  4 lpf.dw
    //  5 lsa.ln.g 6 lsa.ln.b 7..14 lsa.mha {wq,wk,wv,bq,bk,bv,wo,bo}
    // 15 mix_w 16 mix_b 17 proj_w 18 proj_b
    // 19 lnq.g 20 lnq.b 21 lnkv.g 22 lnkv.b 23..30 rp.mha {wq..bo}
    // 31 ffn_ln.g 32 ffn_ln.b 33 w1 34 b1 35 w2 36 b2
    auto P = [&](int l, int off) -> const float* { return (const float*)d_in[7 + l * 37 + off]; };

    uint8_t* wp = (uint8_t*)d_ws;
    auto alloc = [&](size_t bytes) -> void* {
        void* r = (void*)wp;
        wp += (bytes + 255) & ~(size_t)255;
        return r;
    };
    float*  Ak   = (float*)alloc((size_t)2 * Mc * sizeof(float));
    float*  Ag   = (float*)alloc((size_t)Mc * sizeof(float));
    float*  Hg   = (float*)alloc((size_t)Mc * sizeof(float));
    float*  cB   = (float*)alloc((size_t)Mc * sizeof(float));
    float*  gB   = (float*)alloc((size_t)Mc * sizeof(float));
    float*  memF = (float*)alloc((size_t)Mc * Dc * sizeof(float));
    float*  yA   = (float*)alloc((size_t)Mc * Dc * sizeof(float));
    float*  yB   = (float*)alloc((size_t)Mc * Dc * sizeof(float));
    float*  qb   = (float*)alloc((size_t)Mc * Dc * sizeof(float));
    float*  kb   = (float*)alloc((size_t)Mc * Dc * sizeof(float));
    float*  vb   = (float*)alloc((size_t)Mc * Dc * sizeof(float));
    float*  ob   = (float*)alloc((size_t)Mc * Dc * sizeof(float));
    float*  yr   = (float*)alloc((size_t)Mc * Dc * sizeof(float));
    __bf16* aBF  = (__bf16*)alloc((size_t)Mc * Fc * sizeof(__bf16));
    __bf16* aBF2 = (__bf16*)alloc((size_t)Mc * Fc * sizeof(__bf16));
    __bf16* wT   = (__bf16*)alloc((size_t)Fc * Dc * sizeof(__bf16));

    const int ND = Mc * Dc;
    auto gemm = [&](const __bf16* Aact, const float* Wf, int Kd, int Nd,
                    const float* bias, const float* res, float* outF, __bf16* outB, int act) {
        int nt = Kd * Nd;
        k_transpose_bf16<<<(nt + 255) / 256, 256, 0, stream>>>(Wf, wT, Kd, Nd);
        dim3 g(Nd / 64, Mc / 128);
        k_gemm_bf16<<<g, 256, 0, stream>>>(Aact, wT, bias, res, outF, outB, Mc, Nd, Kd, act);
    };

    // --- CTC guidance stats ---
    k_ctc_stats<<<Mc, 256, 0, stream>>>(logits, blank, Ak, Ag, Hg);

    // --- mem = x_m @ inp_w + inp_b ---
    k_f32_to_bf16<<<(ND + 255) / 256, 256, 0, stream>>>(x_m, aBF, ND);
    gemm(aBF, inp_w, Dc, Dc, inp_b, nullptr, memF, nullptr, 0);
    hipMemcpyAsync(yA, memF, (size_t)ND * sizeof(float), hipMemcpyDeviceToDevice, stream);

    float* y = yA;
    float* yAlt = yB;
    for (int l = 0; l < 2; ++l) {
        // DynamicLPF
        k_lpf_coeff<<<Mc, 256, 0, stream>>>(Ag, Hg, P(l,0), P(l,1), P(l,2), P(l,3), cB);
        k_lpf_apply<<<(ND + 255) / 256, 256, 0, stream>>>(y, P(l,4), cB, yAlt, ND);
        { float* t = y; y = yAlt; yAlt = t; }

        // LocalSelfAttn (pre-LN, band 16, residual fused in wo-GEMM)
        k_layernorm<<<Mc / 8, 256, 0, stream>>>(y, P(l,5), P(l,6), nullptr, aBF);
        gemm(aBF, P(l,7), Dc, Dc, P(l,10), nullptr, qb, nullptr, 0);
        gemm(aBF, P(l,8), Dc, Dc, P(l,11), nullptr, kb, nullptr, 0);
        gemm(aBF, P(l,9), Dc, Dc, P(l,12), nullptr, vb, nullptr, 0);
        k_banded_attn<<<(Bc * Hc * Tc) / 8, 256, 0, stream>>>(qb, kb, vb, ob, 16, 0.125f);
        k_f32_to_bf16<<<(ND + 255) / 256, 256, 0, stream>>>(ob, aBF, ND);
        gemm(aBF, P(l,13), Dc, Dc, P(l,14), y, y, nullptr, 0);

        // SoftSpeakerRouter
        k_gate<<<(Mc + 255) / 256, 256, 0, stream>>>(Ak, P(l,15), P(l,16), gB, Mc);
        k_scale_gate<<<(ND + 255) / 256, 256, 0, stream>>>(y, gB, aBF, ND);
        gemm(aBF, P(l,17), Dc, Dc, P(l,18), nullptr, yAlt, nullptr, 0);
        { float* t = y; y = yAlt; yAlt = t; }

        // CrossRepair: banded (24) cross-attn to mem, then FFN
        k_layernorm<<<Mc / 8, 256, 0, stream>>>(y,    P(l,19), P(l,20), nullptr, aBF);   // qq
        k_layernorm<<<Mc / 8, 256, 0, stream>>>(memF, P(l,21), P(l,22), nullptr, aBF2);  // kv
        gemm(aBF,  P(l,23), Dc, Dc, P(l,26), nullptr, qb, nullptr, 0);
        gemm(aBF2, P(l,24), Dc, Dc, P(l,27), nullptr, kb, nullptr, 0);
        gemm(aBF2, P(l,25), Dc, Dc, P(l,28), nullptr, vb, nullptr, 0);
        k_banded_attn<<<(Bc * Hc * Tc) / 8, 256, 0, stream>>>(qb, kb, vb, ob, 24, 0.125f);
        k_f32_to_bf16<<<(ND + 255) / 256, 256, 0, stream>>>(ob, aBF, ND);
        gemm(aBF, P(l,29), Dc, Dc, P(l,30), y, yr, nullptr, 0);                          // yr = y + o@wo
        k_layernorm<<<Mc / 8, 256, 0, stream>>>(yr, P(l,31), P(l,32), nullptr, aBF);     // h
        gemm(aBF,  P(l,33), Dc, Fc, P(l,34), nullptr, nullptr, aBF2, 1);                 // gelu(h@w1+b1) -> bf16
        gemm(aBF2, P(l,35), Fc, Dc, P(l,36), yr, y, nullptr, 0);                         // y = yr + .@w2+b2
    }

    // --- out_ln -> d_out (fp32) ---
    k_layernorm<<<Mc / 8, 256, 0, stream>>>(y, oln_g, oln_b, (float*)d_out, nullptr);
}
